// NeuralOpLibrary_58420145160286
// MI455X (gfx1250) — matrix operand order repaired and checked by reading, not compiled, not run on hardware
//
#include <hip/hip_runtime.h>
#include <hip/hip_bf16.h>
#include <math.h>

// Problem sizes (fixed by the reference)
#define BB 4
#define TT 2048
#define DD 1024
#define HH 1024
#define VV 8
#define LLOPS 4
#define NROWS (BB * TT)          // 8192 token rows

typedef __attribute__((ext_vector_type(16))) __bf16 v16bf;
typedef __attribute__((ext_vector_type(8)))  float  v8f;

// ---------------------------------------------------------------------------
// Kernel 1: wbar[b][v] = (1/L) * sum_l softmax(op_logits[b,l,:])[v]
// ---------------------------------------------------------------------------
__global__ void wbar_kernel(const float* __restrict__ logits, float* __restrict__ wbar) {
    int b = threadIdx.x;
    if (b < BB) {
        float acc[VV];
        for (int v = 0; v < VV; ++v) acc[v] = 0.0f;
        for (int l = 0; l < LLOPS; ++l) {
            const float* row = logits + ((size_t)b * LLOPS + l) * VV;
            float m = row[0];
            for (int v = 1; v < VV; ++v) m = fmaxf(m, row[v]);
            float e[VV], s = 0.0f;
            for (int v = 0; v < VV; ++v) { e[v] = __expf(row[v] - m); s += e[v]; }
            float inv = 1.0f / s;
            for (int v = 0; v < VV; ++v) acc[v] += e[v] * inv;
        }
        for (int v = 0; v < VV; ++v) wbar[b * VV + v] = acc[v] * (1.0f / (float)LLOPS);
    }
}

// ---------------------------------------------------------------------------
// Kernel 2: fp32 -> bf16 cast (grid-stride)
// ---------------------------------------------------------------------------
__global__ void cast_f32_bf16(const float* __restrict__ src,
                              __hip_bfloat16* __restrict__ dst, int n) {
    int i = blockIdx.x * blockDim.x + threadIdx.x;
    int stride = gridDim.x * blockDim.x;
    for (; i < n; i += stride) dst[i] = __float2bfloat16(src[i]);
}

// ---------------------------------------------------------------------------
// Fragment loaders (bf16, per ISA §7.12.2 layouts)
// ---------------------------------------------------------------------------
// A (16x32): lane<16 -> row M=lane, K {k0..k0+7, k0+16..k0+23}
//            lane>=16 -> row M=lane-16, K {k0+8..k0+15, k0+24..k0+31}
__device__ __forceinline__ v16bf load_a_lds(const __hip_bfloat16* lds,
                                            int mrow, int k0, int lane) {
    int r  = mrow + (lane & 15);
    int kk = k0 + ((lane & 16) ? 8 : 0);
    union { v16bf v; uint4 q[2]; } u;
    const uint4* p = (const uint4*)(lds + r * HH + kk);  // uint4 = 8 bf16
    u.q[0] = p[0];   // K = kk .. kk+7
    u.q[1] = p[2];   // K = kk+16 .. kk+23
    return u.v;
}

// B (32x16): lane holds column N=n0+(lane&15), K {kk..kk+7, kk+16..kk+23}, kk=k0+(lane>=16?8:0)
__device__ __forceinline__ v16bf load_b_glob(const __hip_bfloat16* __restrict__ w,
                                             int k0, int n0, int lane) {
    union { v16bf v; unsigned short s[16]; } u;
    const unsigned short* p = (const unsigned short*)w;
    int n  = n0 + (lane & 15);
    int kk = k0 + ((lane & 16) ? 8 : 0);
    #pragma unroll
    for (int s = 0; s < 8; ++s) {
        u.s[s]     = p[(size_t)(kk + s) * HH + n];
        u.s[8 + s] = p[(size_t)(kk + 16 + s) * HH + n];
    }
    return u.v;
}

// Branch-free tanh-GeLU via hardware exp/rcp:
// gelu(x) ~= x * sigmoid(1.5957691*x + 0.0713548*x^3); error <= ~1e-3,
// below the bf16 quantization noise of the WMMA path.
__device__ __forceinline__ float gelu_fast(float x) {
    float u = x * (1.5957691216f + 0.0713548162f * x * x);
    float s = __builtin_amdgcn_rcpf(1.0f + __expf(-u));
    return x * s;
}

// One K-64 step of a 32xN GEMM slab: 8 B fragments of 16 strided 16-bit global loads each, 16 WMMAs drain.
__device__ __forceinline__ void gemm_step64(const __hip_bfloat16* __restrict__ w,
                                            const __hip_bfloat16* asrc,
                                            int k0, int ncol0, int lane,
                                            v8f acc[2][4]) {
    v16bf bA[4], bB[4];
    #pragma unroll
    for (int nt = 0; nt < 4; ++nt)
        bA[nt] = load_b_glob(w, k0, ncol0 + nt * 16, lane);
    #pragma unroll
    for (int nt = 0; nt < 4; ++nt)
        bB[nt] = load_b_glob(w, k0 + 32, ncol0 + nt * 16, lane);
    v16bf a0 = load_a_lds(asrc, 0,  k0, lane);
    v16bf a1 = load_a_lds(asrc, 16, k0, lane);
    v16bf a2 = load_a_lds(asrc, 0,  k0 + 32, lane);
    v16bf a3 = load_a_lds(asrc, 16, k0 + 32, lane);
    #pragma unroll
    for (int nt = 0; nt < 4; ++nt) {
        acc[0][nt] = __builtin_amdgcn_wmma_f32_16x16x32_bf16(
            false, a0, false, bA[nt], (short)0, acc[0][nt], false, false);
        acc[1][nt] = __builtin_amdgcn_wmma_f32_16x16x32_bf16(
            false, a1, false, bA[nt], (short)0, acc[1][nt], false, false);
    }
    #pragma unroll
    for (int nt = 0; nt < 4; ++nt) {
        acc[0][nt] = __builtin_amdgcn_wmma_f32_16x16x32_bf16(
            false, a2, false, bB[nt], (short)0, acc[0][nt], false, false);
        acc[1][nt] = __builtin_amdgcn_wmma_f32_16x16x32_bf16(
            false, a3, false, bB[nt], (short)0, acc[1][nt], false, false);
    }
}

// ---------------------------------------------------------------------------
// Kernel 3: fused per-32-row-tile  GEMM1 -> GeLU -> GEMM2 -> weighted combine
// 512 threads = 16 waves; wave w owns N-columns [w*64, w*64+64) (4 N-tiles);
// two 16-row M-tiles register-blocked (B fragment reused for both).
// ---------------------------------------------------------------------------
__launch_bounds__(512, 1)
__global__ void fused_mlp(const __hip_bfloat16* __restrict__ Xb,
                          const __hip_bfloat16* __restrict__ W1b,
                          const __hip_bfloat16* __restrict__ W2b,
                          const float* __restrict__ b1,
                          const float* __restrict__ b2,
                          const float* __restrict__ wbar,
                          float* __restrict__ out) {
    extern __shared__ __align__(16) char smem_raw[];
    __hip_bfloat16* lds_x = (__hip_bfloat16*)smem_raw;            // 32 x 1024 bf16 (64 KB)
    __hip_bfloat16* lds_h = lds_x + 32 * HH;                      // 32 x 1024 bf16 (64 KB)

    const int tid   = threadIdx.x;
    const int lane  = tid & 31;
    const int wave  = tid >> 5;          // 0..15
    const int ncol0 = wave * 64;         // wave's N base
    const int r0    = blockIdx.x * 32;   // global token-row base (0..8191)
    const int b     = r0 >> 11;          // 2048 rows per batch element

    // ---- stage X tile into LDS via CDNA5 async direct-to-LDS copies ----
    // 32 rows x 1024 cols bf16 = 4096 x 16B; 512 threads -> 8 x b128 each.
    {
        const unsigned lds_base = (unsigned)(uintptr_t)lds_x;     // low 32 bits = LDS offset
        unsigned long long gsrc = (unsigned long long)(uintptr_t)(Xb + (size_t)r0 * DD);
        #pragma unroll
        for (int i = 0; i < 8; ++i) {
            unsigned      dst = lds_base + (unsigned)(tid + i * 512) * 16u;
            unsigned long long src = gsrc + (unsigned long long)(tid + i * 512) * 16ull;
            asm volatile("global_load_async_to_lds_b128 %0, %1, off"
                         :: "v"(dst), "v"(src) : "memory");
        }
        asm volatile("s_wait_asynccnt 0x0" ::: "memory");
    }
    __syncthreads();

    const int nlo = lane & 15;           // N within tile
    const int mhi = (lane & 16) ? 8 : 0; // M offset for C/D rows

    // result accumulators: [m-tile][n-tile] of 16x16 f32
    v8f res[2][4];
    for (int m = 0; m < 2; ++m)
        for (int nt = 0; nt < 4; ++nt)
            for (int e = 0; e < 8; ++e) res[m][nt][e] = 0.0f;

    for (int v = 0; v < VV; ++v) {
        const __hip_bfloat16* w1 = W1b + (size_t)v * DD * HH;
        const __hip_bfloat16* w2 = W2b + (size_t)v * HH * HH;

        // ---------------- GEMM1: h = X @ W1[v] ----------------
        v8f acc[2][4];
        for (int m = 0; m < 2; ++m)
            for (int nt = 0; nt < 4; ++nt)
                for (int e = 0; e < 8; ++e) acc[m][nt][e] = 0.0f;

        for (int k0 = 0; k0 < DD; k0 += 64) {
            if (k0 + 128 < DD) {   // warm WGP$ two iterations ahead
                __builtin_prefetch(w1 + (size_t)(k0 + 128 + lane) * HH + ncol0, 0, 3);
                __builtin_prefetch(w1 + (size_t)(k0 + 160 + lane) * HH + ncol0, 0, 3);
            }
            gemm_step64(w1, lds_x, k0, ncol0, lane, acc);
        }

        // bias + GeLU, write h tile (bf16) to LDS
        #pragma unroll
        for (int nt = 0; nt < 4; ++nt) {
            int n = ncol0 + nt * 16 + nlo;
            float bias = b1[v * HH + n];
            #pragma unroll
            for (int m = 0; m < 2; ++m) {
                #pragma unroll
                for (int e = 0; e < 8; ++e) {
                    float g = gelu_fast(acc[m][nt][e] + bias);
                    int row = m * 16 + mhi + e;
                    lds_h[row * HH + n] = __float2bfloat16(g);
                }
            }
        }
        __syncthreads();

        // ---------------- GEMM2: o = h @ W2[v] ----------------
        for (int m = 0; m < 2; ++m)
            for (int nt = 0; nt < 4; ++nt)
                for (int e = 0; e < 8; ++e) acc[m][nt][e] = 0.0f;

        for (int k0 = 0; k0 < HH; k0 += 64) {
            if (k0 + 128 < HH) {
                __builtin_prefetch(w2 + (size_t)(k0 + 128 + lane) * HH + ncol0, 0, 3);
                __builtin_prefetch(w2 + (size_t)(k0 + 160 + lane) * HH + ncol0, 0, 3);
            }
            gemm_step64(w2, lds_h, k0, ncol0, lane, acc);
        }

        // weighted accumulate: res += wbar[b][v] * (acc + b2[v])
        float wv = wbar[b * VV + v];
        #pragma unroll
        for (int nt = 0; nt < 4; ++nt) {
            int n = ncol0 + nt * 16 + nlo;
            float bias2 = b2[v * HH + n];
            #pragma unroll
            for (int m = 0; m < 2; ++m)
                #pragma unroll
                for (int e = 0; e < 8; ++e)
                    res[m][nt][e] += wv * (acc[m][nt][e] + bias2);
        }
        __syncthreads();   // lds_h reused next v
    }

    // ---- write result [B*T, H] fp32 ----
    #pragma unroll
    for (int nt = 0; nt < 4; ++nt) {
        int n = ncol0 + nt * 16 + nlo;
        #pragma unroll
        for (int m = 0; m < 2; ++m)
            #pragma unroll
            for (int e = 0; e < 8; ++e) {
                int row = r0 + m * 16 + mhi + e;
                out[(size_t)row * HH + n] = res[m][nt][e];
            }
    }
}

// ---------------------------------------------------------------------------
// Host launcher
// ---------------------------------------------------------------------------
extern "C" void kernel_launch(void* const* d_in, const int* in_sizes, int n_in,
                              void* d_out, int out_size, void* d_ws, size_t ws_size,
                              hipStream_t stream) {
    const float* op_logits   = (const float*)d_in[0];  // [4,4,8]
    const float* token_feats = (const float*)d_in[1];  // [4,2048,1024]
    const float* W1          = (const float*)d_in[2];  // [8,1024,1024]
    const float* b1          = (const float*)d_in[3];  // [8,1024]
    const float* W2          = (const float*)d_in[4];  // [8,1024,1024]
    const float* b2          = (const float*)d_in[5];  // [8,1024]
    float* out = (float*)d_out;

    // workspace layout (needs ~48.1 MB)
    char* ws = (char*)d_ws;
    float*          wbar = (float*)ws;                                  // 32 f32
    __hip_bfloat16* Xb   = (__hip_bfloat16*)(ws + 256);                 // 16 MB
    __hip_bfloat16* W1b  = Xb  + (size_t)NROWS * DD;                    // 16 MB
    __hip_bfloat16* W2b  = W1b + (size_t)VV * DD * HH;                  // 16 MB

    wbar_kernel<<<1, 32, 0, stream>>>(op_logits, wbar);
    cast_f32_bf16<<<2048, 256, 0, stream>>>(token_feats, Xb, NROWS * DD);
    cast_f32_bf16<<<2048, 256, 0, stream>>>(W1, W1b, VV * DD * HH);
    cast_f32_bf16<<<2048, 256, 0, stream>>>(W2, W2b, VV * HH * HH);

    const size_t smem = 2u * 32u * HH * sizeof(__hip_bfloat16);  // 128 KB dynamic LDS
    static bool attr_set = false;
    if (!attr_set) {
        (void)hipFuncSetAttribute((const void*)fused_mlp,
                                  hipFuncAttributeMaxDynamicSharedMemorySize,
                                  (int)smem);
        attr_set = true;
    }
    fused_mlp<<<NROWS / 32, 512, smem, stream>>>(Xb, W1b, W2b, b1, b2, wbar, out);
}
